// GatedNNMF_41085657153706
// MI455X (gfx1250) — compile-verified, matrix operand order checked by reading, not run on hardware
//
#include <hip/hip_runtime.h>
#include <math.h>

// ---------------------------------------------------------------------------
// GatedNNMF for MI455X (gfx1250): bf16 WMMA (v_wmma_f32_16x16x32_bf16) for all
// large matmuls, f32 state for the NMF multiplicative updates.
// Software-pipelined: tile k+1 global loads (b128) issued before the WMMA
// block of tile k; staging loops have compile-time trip counts (no exec-mask
// divergence).  All problem dims are exact multiples of the tiles.
// ---------------------------------------------------------------------------

typedef __attribute__((ext_vector_type(16))) __bf16 bf16x16;
typedef __attribute__((ext_vector_type(8)))  __bf16 bf16x8;
typedef __attribute__((ext_vector_type(4)))  __bf16 bf16x4;
typedef __attribute__((ext_vector_type(8)))  float  f32x8;

union AFrag { bf16x16 v; bf16x8 h[2]; };

#define EPI_GELU_SPLIT 0
#define EPI_F32        1
#define EPI_MUL_BF16   2
#define EPI_BIAS_F32   3

// ---------------------------------------------------------------------------
// C[M,N] = epi( sum_k A[m,k]*B[k,n] )
//  ATR: A stored K-major (A[m][k] at A + k*lda + m)  else row-major (m*lda+k)
//  BTR: B stored N-major rows over K (B[k][n] at B + n*ldb + k)  else k*ldb+n
//  CDNA5 ISA 7.12.2 fragment layouts; wave32; 8 waves / 256-thread block.
// ---------------------------------------------------------------------------
template<typename TA, typename TB, bool ATR, bool BTR,
         int TM, int TN, int WR, int WC, int EPI>
__global__ __launch_bounds__(256) void gemm_wmma(
    const TA* __restrict__ A, const TB* __restrict__ B,
    long long lda, long long ldb,
    int M, int N, int K,
    long long strideA, long long strideB, long long strideC,
    const float* __restrict__ bias,
    float* __restrict__ outF,
    __bf16* __restrict__ outH, __bf16* __restrict__ outH2,
    long long ldc)
{
    static_assert(WR * WC == 8, "8 waves per 256-thread block (wave32)");
    constexpr int TK  = 32;            // K per WMMA (bf16 16x16x32)
    constexpr int SA  = TK + 8;        // LDS row stride (bf16) w/ pad
    constexpr int WTM = TM / WR, WTN = TN / WC;
    constexpr int MF  = WTM / 16, NF = WTN / 16;
    constexpr int NT  = 256;

    // staging iteration counts (compile-time; all exact multiples of NT)
    constexpr bool A_F32 = (!ATR && sizeof(TA) == 4);
    constexpr bool A_BF  = (!ATR && sizeof(TA) == 2);
    constexpr int ITAf = A_F32 ? (TM * (TK / 4)) / NT : 1;   // float4 loads
    constexpr int ITAh = A_BF  ? (TM * (TK / 8)) / NT : 1;   // bf16x8 loads
    constexpr int ITAt = ATR   ? ((TM / 8) * TK) / NT : 1;   // bf16x8 along m
    constexpr int ITBt = BTR   ? (TN * (TK / 4)) / NT : 1;   // float4 loads
    constexpr int ITBn = !BTR  ? (TK * (TN / 4)) / NT : 1;   // float4 along n

    __shared__ __align__(16) __bf16 ldsA[TM * SA];
    __shared__ __align__(16) __bf16 ldsB[TN * SA];

    const int tid  = threadIdx.x;
    const int lane = tid & 31, wave = tid >> 5;
    const int half = lane >> 4, r16 = lane & 15;
    const int waveM = (wave / WC) * WTM;
    const int waveN = (wave % WC) * WTN;
    const int tileM = blockIdx.y * TM;
    const int tileN = blockIdx.x * TN;

    if (strideA) A += (size_t)blockIdx.z * strideA;
    if (strideB) B += (size_t)blockIdx.z * strideB;
    if (strideC) {
        if (outF)  outF  += (size_t)blockIdx.z * strideC;
        if (outH)  outH  += (size_t)blockIdx.z * strideC;
        if (outH2) outH2 += (size_t)blockIdx.z * strideC;
    }

    // register staging buffers (unused variants collapse to size 1)
    float4 rAf[ITAf]; bf16x8 rAh[ITAh]; bf16x8 rAt[ITAt];
    float4 rBt[ITBt]; float4 rBn[ITBn];

    auto loadA = [&](int k0) {
        if constexpr (A_F32) {
            #pragma unroll
            for (int it = 0; it < ITAf; ++it) {
                int v = tid + it * NT, m = v >> 3, kc = v & 7;
                rAf[it] = *(const float4*)(A + (size_t)(tileM + m) * lda + k0 + kc * 4);
            }
        } else if constexpr (A_BF) {
            #pragma unroll
            for (int it = 0; it < ITAh; ++it) {
                int v = tid + it * NT, m = v >> 2, kc = v & 3;
                rAh[it] = *(const bf16x8*)(A + (size_t)(tileM + m) * lda + k0 + kc * 8);
            }
        } else {  // ATR: K-major bf16 (xm^T), contiguous over m
            static_assert(!ATR || (sizeof(TA) == 2 && TM == 128), "ATR = bf16/TM=128");
            #pragma unroll
            for (int it = 0; it < ITAt; ++it) {
                int v = tid + it * NT, k = v >> 4, mc = v & 15;   // TM/8 == 16
                rAt[it] = *(const bf16x8*)(A + (size_t)(k0 + k) * lda + tileM + mc * 8);
            }
        }
    };
    auto storeA = [&]() {
        if constexpr (A_F32) {
            #pragma unroll
            for (int it = 0; it < ITAf; ++it) {
                int v = tid + it * NT, m = v >> 3, kc = v & 7;
                bf16x4 o;
                o[0] = (__bf16)rAf[it].x; o[1] = (__bf16)rAf[it].y;
                o[2] = (__bf16)rAf[it].z; o[3] = (__bf16)rAf[it].w;
                *(bf16x4*)&ldsA[m * SA + kc * 4] = o;
            }
        } else if constexpr (A_BF) {
            #pragma unroll
            for (int it = 0; it < ITAh; ++it) {
                int v = tid + it * NT, m = v >> 2, kc = v & 3;
                *(bf16x8*)&ldsA[m * SA + kc * 8] = rAh[it];
            }
        } else {
            #pragma unroll
            for (int it = 0; it < ITAt; ++it) {
                int v = tid + it * NT, k = v >> 4, mc = v & 15;
                #pragma unroll
                for (int j = 0; j < 8; ++j)
                    ldsA[(mc * 8 + j) * SA + k] = rAt[it][j];
            }
        }
    };
    auto loadB = [&](int k0) {
        static_assert(sizeof(TB) == 4, "B is always f32");
        if constexpr (BTR) {
            #pragma unroll
            for (int it = 0; it < ITBt; ++it) {
                int v = tid + it * NT, n = v >> 3, kc = v & 7;
                rBt[it] = *(const float4*)(B + (size_t)(tileN + n) * ldb + k0 + kc * 4);
            }
        } else {
            #pragma unroll
            for (int it = 0; it < ITBn; ++it) {
                int v = tid + it * NT, k = v / (TN / 4), nc = v % (TN / 4);
                rBn[it] = *(const float4*)(B + (size_t)(k0 + k) * ldb + tileN + nc * 4);
            }
        }
    };
    auto storeB = [&]() {
        if constexpr (BTR) {
            #pragma unroll
            for (int it = 0; it < ITBt; ++it) {
                int v = tid + it * NT, n = v >> 3, kc = v & 7;
                bf16x4 o;
                o[0] = (__bf16)rBt[it].x; o[1] = (__bf16)rBt[it].y;
                o[2] = (__bf16)rBt[it].z; o[3] = (__bf16)rBt[it].w;
                *(bf16x4*)&ldsB[n * SA + kc * 4] = o;
            }
        } else {
            #pragma unroll
            for (int it = 0; it < ITBn; ++it) {
                int v = tid + it * NT, k = v / (TN / 4), nc = v % (TN / 4);
                ldsB[(nc * 4 + 0) * SA + k] = (__bf16)rBn[it].x;
                ldsB[(nc * 4 + 1) * SA + k] = (__bf16)rBn[it].y;
                ldsB[(nc * 4 + 2) * SA + k] = (__bf16)rBn[it].z;
                ldsB[(nc * 4 + 3) * SA + k] = (__bf16)rBn[it].w;
            }
        }
    };

    f32x8 acc[MF][NF];
    const f32x8 zf = {0.f,0.f,0.f,0.f,0.f,0.f,0.f,0.f};
    #pragma unroll
    for (int i = 0; i < MF; ++i)
        #pragma unroll
        for (int j = 0; j < NF; ++j) acc[i][j] = zf;

    // ---------------- software-pipelined main loop ----------------
    loadA(0); loadB(0);
    for (int k0 = 0; k0 < K; k0 += TK) {
        __syncthreads();
        storeA(); storeB();
        __syncthreads();
        const int kn = k0 + TK;
        if (kn < K) {                 // issue next tile's global loads early
            loadA(kn); loadB(kn);
            // prefetch two tiles ahead (global_prefetch_b8)
            if (kn + TK < K) {
                const TA* pa = A + (ATR ? (size_t)(kn + TK) * lda + tileM + (tid & (TM - 1))
                                        : (size_t)(tileM + (tid & (TM - 1))) * lda + (kn + TK));
                __builtin_prefetch(pa, 0, 0);
                const TB* pb = B + (BTR ? (size_t)(tileN + (tid & (TN - 1))) * ldb + (kn + TK)
                                        : (size_t)(kn + TK) * ldb + tileN + (tid & (TN - 1)));
                __builtin_prefetch(pb, 0, 0);
            }
        }

        AFrag a[MF], b[NF];
        #pragma unroll
        for (int i = 0; i < MF; ++i) {
            const __bf16* p = &ldsA[(waveM + i * 16 + r16) * SA + 8 * half];
            a[i].h[0] = *(const bf16x8*)(p);
            a[i].h[1] = *(const bf16x8*)(p + 16);
        }
        #pragma unroll
        for (int j = 0; j < NF; ++j) {
            const __bf16* p = &ldsB[(waveN + j * 16 + r16) * SA + 8 * half];
            b[j].h[0] = *(const bf16x8*)(p);
            b[j].h[1] = *(const bf16x8*)(p + 16);
        }
        #pragma unroll
        for (int i = 0; i < MF; ++i)
            #pragma unroll
            for (int j = 0; j < NF; ++j)
                acc[i][j] = __builtin_amdgcn_wmma_f32_16x16x32_bf16(
                    false, a[i].v, false, b[j].v, (short)0, acc[i][j],
                    false, false);
    }

    // ---- epilogue ----
    #pragma unroll
    for (int i = 0; i < MF; ++i)
        #pragma unroll
        for (int j = 0; j < NF; ++j) {
            int gn = tileN + waveN + j * 16 + r16;
            #pragma unroll
            for (int rr = 0; rr < 8; ++rr) {
                int gm = tileM + waveM + i * 16 + rr + 8 * half;
                float v = acc[i][j][rr];
                if (EPI == EPI_GELU_SPLIT) {
                    v += bias[gn];
                    v = 0.5f * v * (1.0f + erff(v * 0.70710678118654752f));
                    int sp = (int)ldc;                  // split = 2048
                    if (gn < sp) outH [(size_t)gm * sp + gn]        = (__bf16)v;
                    else         outH2[(size_t)gm * sp + (gn - sp)] = (__bf16)v;
                } else if (EPI == EPI_F32) {
                    outF[(size_t)gm * ldc + gn] = v;
                } else if (EPI == EPI_MUL_BF16) {
                    size_t idx = (size_t)gm * ldc + gn;
                    v *= (float)outH2[idx];             // gate by z1 (in-place)
                    outH[idx] = (__bf16)v;
                } else { // EPI_BIAS_F32
                    outF[(size_t)gm * ldc + gn] = v + bias[gn];
                }
            }
        }
}

// ---- LayerNorm(eps=1e-5) + ReLU, in-place on bf16 rows of 2048 ----
__global__ __launch_bounds__(256) void ln_relu_kernel(
    __bf16* __restrict__ z, const float* __restrict__ g,
    const float* __restrict__ b)
{
    const int Nc = 2048;
    __bf16* p = z + (size_t)blockIdx.x * Nc;
    const int t = threadIdx.x;
    bf16x8 d = *(const bf16x8*)(p + t * 8);     // 16B per lane, coalesced
    float vals[8], s = 0.f, s2 = 0.f;
    #pragma unroll
    for (int j = 0; j < 8; ++j) {
        float v = (float)d[j];
        vals[j] = v; s += v; s2 += v * v;
    }
    __shared__ float rs[256], rq[256];
    rs[t] = s; rq[t] = s2;
    __syncthreads();
    for (int off = 128; off > 0; off >>= 1) {
        if (t < off) { rs[t] += rs[t + off]; rq[t] += rq[t + off]; }
        __syncthreads();
    }
    float mu   = rs[0] / Nc;
    float var  = rq[0] / Nc - mu * mu;
    float rstd = rsqrtf(var + 1e-5f);
    bf16x8 o;
    #pragma unroll
    for (int j = 0; j < 8; ++j) {
        int idx = t * 8 + j;
        float v = (vals[j] - mu) * rstd * g[idx] + b[idx];
        o[j] = (__bf16)fmaxf(v, 0.f);
    }
    *(bf16x8*)(p + t * 8) = o;
}

// ---- softmax over R=64 (one thread per (b,n) row) ----
__global__ void softmax_r_kernel(const float* __restrict__ num,
                                 float* __restrict__ coef, int rows)
{
    int row = blockIdx.x * blockDim.x + threadIdx.x;
    if (row >= rows) return;
    const float* p = num + (size_t)row * 64;
    float* q = coef + (size_t)row * 64;
    float m = -1e30f;
    for (int i = 0; i < 64; ++i) m = fmaxf(m, p[i]);
    float s = 0.f;
    for (int i = 0; i < 64; ++i) s += __expf(p[i] - m);
    float inv = 1.f / s;
    for (int i = 0; i < 64; ++i) q[i] = __expf(p[i] - m) * inv;
}

// ---- broadcast bases0 (1,D,R) -> (B,D,R) ----
__global__ void bcast_kernel(const float* __restrict__ src,
                             float* __restrict__ dst, int per, int total)
{
    int i = blockIdx.x * 256 + threadIdx.x;
    if (i < total) dst[i] = src[i % per];
}

__global__ void zero_kernel(float* p, int n)
{
    int i = blockIdx.x * 256 + threadIdx.x;
    if (i < n) p[i] = 0.f;
}

// ---- 64x64 Gram: C[r,s] += sum_k X[k,r]*X[k,s], split-K + atomics ----
__global__ __launch_bounds__(256) void gram64_kernel(
    const float* __restrict__ X, float* __restrict__ C, int K, int kchunk)
{
    const float* Xb = X + (size_t)blockIdx.y * K * 64;
    float* Cb = C + (size_t)blockIdx.y * 4096;
    const int t = threadIdx.x;
    const int r = t >> 2, s0 = (t & 3) * 16;
    const int k0 = blockIdx.x * kchunk;
    float acc[16];
    #pragma unroll
    for (int j = 0; j < 16; ++j) acc[j] = 0.f;
    __shared__ float rows[16][64];
    for (int kk = k0; kk < k0 + kchunk; kk += 16) {
        __syncthreads();
        #pragma unroll
        for (int j = 0; j < 4; ++j) {
            int idx = t + j * 256;               // 0..1023
            rows[idx >> 6][idx & 63] = Xb[(size_t)(kk + (idx >> 6)) * 64 + (idx & 63)];
        }
        __syncthreads();
        #pragma unroll
        for (int kr = 0; kr < 16; ++kr) {
            float xv = rows[kr][r];
            #pragma unroll
            for (int j = 0; j < 16; ++j) acc[j] += xv * rows[kr][s0 + j];
        }
    }
    #pragma unroll
    for (int j = 0; j < 16; ++j) atomicAdd(&Cb[r * 64 + s0 + j], acc[j]);
}

// ---- multiplicative update: out[row,t] = in*num / (sum_s in[row,s]*G[s,t] + eps)
__global__ __launch_bounds__(64) void mult_update_kernel(
    const float* __restrict__ in, const float* __restrict__ num,
    const float* __restrict__ G, float* __restrict__ out)
{
    const int bz = blockIdx.y, t = threadIdx.x;
    const size_t rowoff = ((size_t)bz * gridDim.x + blockIdx.x) * 64;
    const float* Gb = G + (size_t)bz * 4096;
    __shared__ float row[64];
    float c0 = in[rowoff + t];
    float nv = num[rowoff + t];
    row[t] = c0;
    __syncthreads();
    float dot = 0.f;
    for (int s = 0; s < 64; ++s) dot += row[s] * Gb[s * 64 + t];
    out[rowoff + t] = c0 * nv / (dot + 1e-6f);
}

// ---------------------------------------------------------------------------
extern "C" void kernel_launch(void* const* d_in, const int* in_sizes, int n_in,
                              void* d_out, int out_size, void* d_ws, size_t ws_size,
                              hipStream_t stream) {
    (void)in_sizes; (void)n_in; (void)out_size; (void)ws_size;
    const float* x      = (const float*)d_in[0];
    const float* U_w    = (const float*)d_in[1];
    const float* U_b    = (const float*)d_in[2];
    const float* ln_g   = (const float*)d_in[3];
    const float* ln_b   = (const float*)d_in[4];
    const float* bases0 = (const float*)d_in[5];
    const float* V_w    = (const float*)d_in[6];
    const float* V_b    = (const float*)d_in[7];
    float* out = (float*)d_out;

    constexpr int Bb = 8, F = 1024, FFN = 4096, Nc = 2048, Rr = 64;
    constexpr int D = 1024, Nn = 2048, Mr = 8192;   // Mr = B*S
    constexpr int STEPS = 6;

    char* w = (char*)d_ws;
    auto alloc = [&](size_t bytes) {
        char* p = w; w += (bytes + 255) & ~(size_t)255; return p;
    };
    __bf16* z1  = (__bf16*)alloc((size_t)Mr * Nc * 2);   // z1 (later gated g)
    __bf16* xm  = (__bf16*)alloc((size_t)Mr * Nc * 2);   // z2raw -> xm
    float* basesA = (float*)alloc((size_t)Bb * D * Rr * 4);
    float* basesB = (float*)alloc((size_t)Bb * D * Rr * 4);
    float* coefA  = (float*)alloc((size_t)Bb * Nn * Rr * 4);
    float* coefB  = (float*)alloc((size_t)Bb * Nn * Rr * 4);
    float* num    = (float*)alloc((size_t)Bb * Nn * Rr * 4);  // reused for num2
    float* gram   = (float*)alloc((size_t)Bb * Rr * Rr * 4);  // btb / ctc

    dim3 blk(256);

    // G1: h = gelu(x @ U_w^T + U_b), split -> z1 (bf16), z2raw (bf16, in xm)
    gemm_wmma<float, float, false, true, 128, 128, 2, 4, EPI_GELU_SPLIT>
        <<<dim3(FFN / 128, Mr / 128, 1), blk, 0, stream>>>(
            x, U_w, F, F, Mr, FFN, F, 0, 0, 0,
            U_b, nullptr, z1, xm, Nc);

    // LayerNorm + ReLU in place -> xm
    ln_relu_kernel<<<dim3(Mr), blk, 0, stream>>>(xm, ln_g, ln_b);

    // broadcast bases0 -> basesA
    bcast_kernel<<<dim3((Bb * D * Rr + 255) / 256), blk, 0, stream>>>(
        bases0, basesA, D * Rr, Bb * D * Rr);

    float* bc = basesA; float* bn = basesB;
    float* cc = coefA;  float* cn = coefB;

    auto launch_num = [&](const float* bases_p) {  // num = xm^T @ bases (B,N,R)
        gemm_wmma<__bf16, float, true, false, 128, 64, 4, 2, EPI_F32>
            <<<dim3(1, Nn / 128, Bb), blk, 0, stream>>>(
                xm, bases_p, Nn, Rr, Nn, Rr, D,
                (long long)D * Nn, (long long)D * Rr, (long long)Nn * Rr,
                nullptr, num, nullptr, nullptr, Rr);
    };
    auto launch_gram = [&](const float* X, int K) {
        zero_kernel<<<dim3((Bb * 4096 + 255) / 256), blk, 0, stream>>>(gram, Bb * 4096);
        gram64_kernel<<<dim3(K / 128, Bb), blk, 0, stream>>>(X, gram, K, 128);
    };

    // coef init: softmax(xm^T @ bases)
    launch_num(bc);
    softmax_r_kernel<<<dim3((Bb * Nn + 255) / 256), blk, 0, stream>>>(num, cc, Bb * Nn);

    for (int it = 0; it < STEPS; ++it) {
        // coef update
        launch_num(bc);
        launch_gram(bc, D);                                   // btb
        mult_update_kernel<<<dim3(Nn, Bb), dim3(64), 0, stream>>>(cc, num, gram, cn);
        { float* t = cc; cc = cn; cn = t; }
        // bases update: num2 = xm @ coef (B,D,R)
        gemm_wmma<__bf16, float, false, false, 128, 64, 4, 2, EPI_F32>
            <<<dim3(1, D / 128, Bb), blk, 0, stream>>>(
                xm, cc, Nn, Rr, D, Rr, Nn,
                (long long)D * Nn, (long long)Nn * Rr, (long long)D * Rr,
                nullptr, num, nullptr, nullptr, Rr);
        launch_gram(cc, Nn);                                  // ctc
        mult_update_kernel<<<dim3(D, Bb), dim3(64), 0, stream>>>(bc, num, gram, bn);
        { float* t = bc; bc = bn; bn = t; }
    }

    // compute_coef: one more coef update
    launch_num(bc);
    launch_gram(bc, D);
    mult_update_kernel<<<dim3(Nn, Bb), dim3(64), 0, stream>>>(cc, num, gram, cn);
    { float* t = cc; cc = cn; cn = t; }

    // reconstruct + gate: g = z1 * (bases @ coef^T), in place over z1
    gemm_wmma<float, float, false, true, 128, 128, 2, 4, EPI_MUL_BF16>
        <<<dim3(Nn / 128, D / 128, Bb), blk, 0, stream>>>(
            bc, cc, Rr, Rr, D, Nn, Rr,
            (long long)D * Rr, (long long)Nn * Rr, (long long)D * Nn,
            nullptr, nullptr, z1, z1, Nc);

    // out = g @ V_w^T + V_b
    gemm_wmma<__bf16, float, false, true, 128, 128, 2, 4, EPI_BIAS_F32>
        <<<dim3(F / 128, Mr / 128, 1), blk, 0, stream>>>(
            z1, V_w, Nc, Nc, Mr, F, Nc, 0, 0, 0,
            V_b, out, nullptr, nullptr, F);
}
